// CloudCrop_85787676770331
// MI455X (gfx1250) — compile-verified, matrix operand order checked by reading, not run on hardware
//
#include <hip/hip_runtime.h>
#include <hip/hip_bf16.h>

#define RADIUS 0.05f
#define HMIN  -0.02f
#define HMAX   0.04f
#define NS     32
#define CIN    512
#define CMID   256
#define COUT   256
#define NB     4
#define NPTS   1024
#define K1PAD  544            // 515 padded to multiple of 32
#define LDK1   552            // LDS leading dim for X  (mult of 8 elems -> 16B rows)
#define LDK2   264            // LDS leading dim for Y1

typedef __attribute__((ext_vector_type(16))) __bf16 v16bf;
typedef __attribute__((ext_vector_type(8)))  __bf16 v8bf;
typedef __attribute__((ext_vector_type(8)))  float  v8f;

union V16U { v16bf v; v8bf h[2]; };

static __device__ __forceinline__ __bf16 f2bf(float f) {
  unsigned u = __builtin_bit_cast(unsigned, f);
  unsigned r = u + 0x7fffu + ((u >> 16) & 1u);   // round-to-nearest-even
  unsigned short hs = (unsigned short)(r >> 16);
  return __builtin_bit_cast(__bf16, hs);
}

// Load one 16x32 (A) or 32x16 (B) bf16 WMMA fragment.
// Memory layout: row-major [16 rows][ld], rows = M (A) or N (B), cols = K.
// ISA layout: lane l<16 -> row l, K {0..7} in v0-3, {16..23} in v4-7;
//             lane l>=16 -> row l-16, K {8..15},{24..31}.
static __device__ __forceinline__ v16bf load_frag(const __bf16* base, int row0,
                                                  int ld, int k0, int lane) {
  int l    = lane & 15;
  int koff = (lane >> 4) << 3;       // 0 or 8
  const __bf16* p = base + (size_t)(row0 + l) * ld + k0 + koff;
  V16U u;
  u.h[0] = *(const v8bf*)(p);        // K koff..koff+7   (16B)
  u.h[1] = *(const v8bf*)(p + 16);   // K koff+16..+23   (16B)
  return u.v;
}

static __device__ __forceinline__ v8f wmma_bf16(v16bf a, v16bf b, v8f c) {
  return __builtin_amdgcn_wmma_f32_16x16x32_bf16(false, a, false, b,
                                                 (short)0, c, false, false);
}

// ---------------- kernel 0: fold BN into bf16 weights ----------------
__global__ __launch_bounds__(256) void prep_kernel(
    const float* __restrict__ w1, const float* __restrict__ g1,
    const float* __restrict__ b1, const float* __restrict__ m1,
    const float* __restrict__ v1,
    const float* __restrict__ w2, const float* __restrict__ g2,
    const float* __restrict__ b2, const float* __restrict__ m2,
    const float* __restrict__ v2,
    __bf16* __restrict__ w1b, __bf16* __restrict__ w2b,
    float* __restrict__ bias1, float* __restrict__ bias2) {
  int o = blockIdx.x;          // 0..255 output channel
  int t = threadIdx.x;         // 0..255
  float s1 = g1[o] * rsqrtf(v1[o] + 1e-5f);
  float s2 = g2[o] * rsqrtf(v2[o] + 1e-5f);
  if (t == 0) {
    bias1[o] = b1[o] - m1[o] * s1;
    bias2[o] = b2[o] - m2[o] * s2;
  }
  for (int c = t; c < K1PAD; c += 256) {
    float w = (c < 515) ? w1[o * 515 + c] * s1 : 0.0f;
    w1b[(size_t)o * K1PAD + c] = f2bf(w);
  }
  w2b[(size_t)o * CMID + t] = f2bf(w2[o * CMID + t] * s2);
}

// ---------------- kernel 1: cylinder query (1 wave32 per query) ----------------
__global__ __launch_bounds__(256) void query_kernel(
    const float* __restrict__ xyz, const float* __restrict__ rot,
    int* __restrict__ idxw, float* __restrict__ relw) {
  int wid  = threadIdx.x >> 5;
  int lane = threadIdx.x & 31;
  int qid  = blockIdx.x * 8 + wid;
  int b    = qid >> 10;
  int n    = qid & 1023;

  __shared__ int sIdx[8][NS];

  const float* R = rot + (size_t)qid * 9;
  float r00 = R[0], r01 = R[1], r02 = R[2];
  float r10 = R[3], r11 = R[4], r12 = R[5];
  float r20 = R[6], r21 = R[7], r22 = R[8];
  const float* X = xyz + (size_t)b * NPTS * 3;
  float cx = X[n * 3 + 0], cy = X[n * 3 + 1], cz = X[n * 3 + 2];

  int count = 0;
  for (int base = 0; base < NPTS && count < NS; base += 32) {
    int m = base + lane;
    float dx = X[m * 3 + 0] - cx;
    float dy = X[m * 3 + 1] - cy;
    float dz = X[m * 3 + 2] - cz;
    float rx = r00 * dx + r01 * dy + r02 * dz;   // rot @ rel
    float ry = r10 * dx + r11 * dy + r12 * dz;
    float rz = r20 * dx + r21 * dy + r22 * dz;
    bool ok = (ry * ry + rz * rz < RADIUS * RADIUS) && (rx > HMIN) && (rx < HMAX);
    unsigned mask = (unsigned)__ballot(ok);
    int pos = count + __popc(mask & ((1u << lane) - 1u));
    if (ok && pos < NS) sIdx[wid][pos] = m;
    count += __popc(mask);
  }
  if (count > NS) count = NS;
  __syncthreads();

  int first = (count > 0) ? sIdx[wid][0] : 0;
  int s  = lane;
  int mi = (s < count) ? sIdx[wid][s] : first;
  idxw[(size_t)qid * NS + s] = mi;

  float dx = X[mi * 3 + 0] - cx;
  float dy = X[mi * 3 + 1] - cy;
  float dz = X[mi * 3 + 2] - cz;
  float inv = 1.0f / RADIUS;
  // second einsum is rel . rot (rot^T applied)
  relw[(size_t)qid * 96 + s * 3 + 0] = (dx * r00 + dy * r10 + dz * r20) * inv;
  relw[(size_t)qid * 96 + s * 3 + 1] = (dx * r01 + dy * r11 + dz * r21) * inv;
  relw[(size_t)qid * 96 + s * 3 + 2] = (dx * r02 + dy * r12 + dz * r22) * inv;
}

// ---------------- kernel 2: fused gather + GEMM1 + GEMM2 + maxpool ----------------
#define SM_XS   0
#define SM_Y1   (32 * LDK1 * 2)
#define SM_B1   (SM_Y1 + 32 * LDK2 * 2)
#define SM_IDX  (SM_B1 + 256 * 4)
#define SM_TOT  (SM_IDX + 32 * 4)

__global__ __launch_bounds__(256) void mlp_kernel(
    const float* __restrict__ feat, const int* __restrict__ idxw,
    const float* __restrict__ relw,
    const __bf16* __restrict__ w1b, const __bf16* __restrict__ w2b,
    const float* __restrict__ bias1, const float* __restrict__ bias2,
    float* __restrict__ out) {
  __shared__ __align__(16) unsigned char smem[SM_TOT];
  __bf16* Xs   = (__bf16*)(smem + SM_XS);    // [32 samples][LDK1] (K-major per sample)
  __bf16* Y1s  = (__bf16*)(smem + SM_Y1);    // [32 samples][LDK2]
  float*  sb1  = (float*)(smem + SM_B1);     // 256
  int*    sidx = (int*)(smem + SM_IDX);      // 32
  float*  Outs = (float*)(smem + SM_XS);     // [256][32] aliases Xs

  int tid = threadIdx.x;
  int qid = blockIdx.x;
  int b = qid >> 10, n = qid & 1023;

  if (tid < 32) sidx[tid] = idxw[(size_t)qid * NS + tid];
  sb1[tid] = bias1[tid];
  __syncthreads();

  // rows 0..2: rotated relative coords
  if (tid < 96) {
    int s = tid / 3, k = tid % 3;
    Xs[s * LDK1 + k] = f2bf(relw[(size_t)qid * 96 + s * 3 + k]);
  }
  // zero K pad cols 515..543
  {
    __bf16 bz = __builtin_bit_cast(__bf16, (unsigned short)0);
    for (int e = tid; e < 29 * 32; e += 256) {
      int s = e / 29, c = e % 29;
      Xs[s * LDK1 + 515 + c] = bz;
    }
  }
  // rows 3..514: gathered features (wave covers all 32 samples of one channel)
  {
    const float* F = feat + (size_t)b * CIN * NPTS;
    for (int e = tid; e < CIN * NS; e += 256) {
      int c = e >> 5, s = e & 31;
      Xs[s * LDK1 + 3 + c] = f2bf(F[(size_t)c * NPTS + sidx[s]]);
    }
  }
  __syncthreads();

  int wid = tid >> 5, lane = tid & 31;
  int m0  = wid * 32;                 // this wave's 32 output rows
  int mh  = (lane >> 4) << 3;         // C/D: lanes 16-31 hold M+8
  int nl  = lane & 15;

  // ---- GEMM1: W1(256x544) x X(544x32) ----
  v8f acc[2][2] = {};
  for (int kt = 0; kt < K1PAD / 32; ++kt) {
    int k0 = kt * 32;
    v16bf fb0 = load_frag(Xs, 0,  LDK1, k0, lane);
    v16bf fb1 = load_frag(Xs, 16, LDK1, k0, lane);
    v16bf fa0 = load_frag(w1b, m0,      K1PAD, k0, lane);
    v16bf fa1 = load_frag(w1b, m0 + 16, K1PAD, k0, lane);
    acc[0][0] = wmma_bf16(fa0, fb0, acc[0][0]);
    acc[0][1] = wmma_bf16(fa0, fb1, acc[0][1]);
    acc[1][0] = wmma_bf16(fa1, fb0, acc[1][0]);
    acc[1][1] = wmma_bf16(fa1, fb1, acc[1][1]);
  }
  // bias + ReLU -> bf16 Y1 (stored sample-major/transposed for GEMM2 B-frags)
#pragma unroll
  for (int mi = 0; mi < 2; ++mi)
#pragma unroll
    for (int nt = 0; nt < 2; ++nt)
#pragma unroll
      for (int r = 0; r < 8; ++r) {
        int m  = m0 + mi * 16 + mh + r;
        int nn = nt * 16 + nl;
        float v = acc[mi][nt][r] + sb1[m];
        v = v > 0.0f ? v : 0.0f;
        Y1s[nn * LDK2 + m] = f2bf(v);
      }
  __syncthreads();

  // ---- GEMM2: W2(256x256) x Y1(256x32) ----
  v8f acc2[2][2] = {};
  for (int kt = 0; kt < CMID / 32; ++kt) {
    int k0 = kt * 32;
    v16bf fb0 = load_frag(Y1s, 0,  LDK2, k0, lane);
    v16bf fb1 = load_frag(Y1s, 16, LDK2, k0, lane);
    v16bf fa0 = load_frag(w2b, m0,      CMID, k0, lane);
    v16bf fa1 = load_frag(w2b, m0 + 16, CMID, k0, lane);
    acc2[0][0] = wmma_bf16(fa0, fb0, acc2[0][0]);
    acc2[0][1] = wmma_bf16(fa0, fb1, acc2[0][1]);
    acc2[1][0] = wmma_bf16(fa1, fb0, acc2[1][0]);
    acc2[1][1] = wmma_bf16(fa1, fb1, acc2[1][1]);
  }
  // raw accumulators to LDS (bias/ReLU commute with max)
#pragma unroll
  for (int mi = 0; mi < 2; ++mi)
#pragma unroll
    for (int nt = 0; nt < 2; ++nt)
#pragma unroll
      for (int r = 0; r < 8; ++r) {
        int m  = m0 + mi * 16 + mh + r;
        int nn = nt * 16 + nl;
        Outs[m * NS + nn] = acc2[mi][nt][r];
      }
  __syncthreads();

  // ---- maxpool over 32 samples, bias2 + ReLU ----
  {
    int o = tid;
    float mv = Outs[o * NS + 0];
#pragma unroll
    for (int s = 1; s < NS; ++s) mv = fmaxf(mv, Outs[o * NS + s]);
    float v = mv + bias2[o];
    out[((size_t)b * COUT + o) * NPTS + n] = v > 0.0f ? v : 0.0f;
  }
}

// ---------------- workspace layout ----------------
#define OFF_W1B 0u
#define OFF_W2B (OFF_W1B + 256u * K1PAD * 2u)        // 278528
#define OFF_B1  (OFF_W2B + 256u * 256u * 2u)         // 409600
#define OFF_B2  (OFF_B1 + 1024u)                     // 410624
#define OFF_IDX (OFF_B2 + 1024u)                     // 411648
#define OFF_REL (OFF_IDX + 4096u * 32u * 4u)         // 935936
// total: OFF_REL + 4096*32*3*4 = 2,508,800 bytes

extern "C" void kernel_launch(void* const* d_in, const int* in_sizes, int n_in,
                              void* d_out, int out_size, void* d_ws, size_t ws_size,
                              hipStream_t stream) {
  const float* xyz  = (const float*)d_in[0];
  const float* feat = (const float*)d_in[1];
  const float* rot  = (const float*)d_in[2];
  const float* w1   = (const float*)d_in[3];
  const float* g1   = (const float*)d_in[4];
  const float* b1   = (const float*)d_in[5];
  const float* m1   = (const float*)d_in[6];
  const float* v1   = (const float*)d_in[7];
  const float* w2   = (const float*)d_in[8];
  const float* g2   = (const float*)d_in[9];
  const float* b2   = (const float*)d_in[10];
  const float* m2   = (const float*)d_in[11];
  const float* v2   = (const float*)d_in[12];
  float* out = (float*)d_out;

  char* ws = (char*)d_ws;
  __bf16* w1b  = (__bf16*)(ws + OFF_W1B);
  __bf16* w2b  = (__bf16*)(ws + OFF_W2B);
  float* bias1 = (float*)(ws + OFF_B1);
  float* bias2 = (float*)(ws + OFF_B2);
  int*   idxw  = (int*)(ws + OFF_IDX);
  float* relw  = (float*)(ws + OFF_REL);

  prep_kernel<<<256, 256, 0, stream>>>(w1, g1, b1, m1, v1,
                                       w2, g2, b2, m2, v2,
                                       w1b, w2b, bias1, bias2);
  query_kernel<<<(NB * NPTS) / 8, 256, 0, stream>>>(xyz, rot, idxw, relw);
  mlp_kernel<<<NB * NPTS, 256, 0, stream>>>(feat, idxw, relw,
                                            w1b, w2b, bias1, bias2, out);
}
